// CasualGraph_77077483094345
// MI455X (gfx1250) — matrix operand order repaired and checked by reading, not compiled, not run on hardware
//
#include <hip/hip_runtime.h>
#include <stdint.h>

// ---------------------------------------------------------------------------
// Types for CDNA5 WMMA (wave32): A/B fragments are v16h, C/D is v8f.
// ---------------------------------------------------------------------------
typedef _Float16 h16 __attribute__((ext_vector_type(16)));
typedef _Float16 h8  __attribute__((ext_vector_type(8)));
typedef float    f8  __attribute__((ext_vector_type(8)));
typedef int      v4i __attribute__((ext_vector_type(4)));

#define NN   8192   // nodes
#define EE   4096   // hyperedges
#define DD   256    // embedding dim
#define KK   8192   // GEMM reduction dim (== NN)

// ---------------------------------------------------------------------------
// Optional CDNA5 async global->LDS path (ASYNCcnt-tracked), guarded so the
// file compiles on toolchains without the builtin (falls back to reg staging).
// Probe-learned signature: (as1 v4i* src, as3 v4i* lds, imm offset, imm cpol)
// ---------------------------------------------------------------------------
#if defined(__gfx1250__) && __has_builtin(__builtin_amdgcn_global_load_async_to_lds_b128)
#define HG_ASYNC_LDS 1
__device__ __forceinline__ void hg_async_b128(const _Float16* g, _Float16* l) {
  __builtin_amdgcn_global_load_async_to_lds_b128(
      (__attribute__((address_space(1))) v4i*)(uintptr_t)g,
      (__attribute__((address_space(3))) v4i*)l,
      0, 0);
}
__device__ __forceinline__ void hg_wait_async0() {
#if __has_builtin(__builtin_amdgcn_s_wait_asynccnt)
  __builtin_amdgcn_s_wait_asynccnt(0);
#else
  asm volatile("s_wait_asynccnt 0" ::: "memory");
#endif
}
#endif

// ---------------------------------------------------------------------------
// Tiled transpose + f32->f16 convert.  in: f32 [R][C] (R,C multiples of 64)
//   outT: f16 [C][R]   (transposed)
//   outS: f16 [R][C]   (straight copy, optional / may be nullptr)
// ---------------------------------------------------------------------------
__global__ __launch_bounds__(256) void hg_transpose_cvt(
    const float* __restrict__ in, _Float16* __restrict__ outT,
    _Float16* __restrict__ outS, int R, int C)
{
  __shared__ _Float16 tile[64][65];
  const int r0 = blockIdx.x * 64;
  const int c0 = blockIdx.y * 64;
  const int tid = threadIdx.x;
  const int tr = tid >> 2;          // 0..63
  const int tc = (tid & 3) * 16;    // 0,16,32,48

  _Float16 hv[16];
#pragma unroll
  for (int i = 0; i < 16; i += 4) {
    float4 v = *(const float4*)(&in[(size_t)(r0 + tr) * C + c0 + tc + i]);
    hv[i + 0] = (_Float16)v.x; hv[i + 1] = (_Float16)v.y;
    hv[i + 2] = (_Float16)v.z; hv[i + 3] = (_Float16)v.w;
  }
#pragma unroll
  for (int i = 0; i < 16; ++i) tile[tr][tc + i] = hv[i];

  if (outS) {
    h8 p0, p1;
#pragma unroll
    for (int i = 0; i < 8; ++i) { p0[i] = hv[i]; p1[i] = hv[8 + i]; }
    *(h8*)(&outS[(size_t)(r0 + tr) * C + c0 + tc])     = p0;
    *(h8*)(&outS[(size_t)(r0 + tr) * C + c0 + tc + 8]) = p1;
  }
  __syncthreads();

  _Float16 tv[16];
#pragma unroll
  for (int i = 0; i < 16; ++i) tv[i] = tile[tc + i][tr];
  h8 q0, q1;
#pragma unroll
  for (int i = 0; i < 8; ++i) { q0[i] = tv[i]; q1[i] = tv[8 + i]; }
  *(h8*)(&outT[(size_t)(c0 + tr) * R + r0 + tc])     = q0;
  *(h8*)(&outT[(size_t)(c0 + tr) * R + r0 + tc + 8]) = q1;
}

// ---------------------------------------------------------------------------
// WMMA GEMM:  C[M][256] (f32) = A[M][8192] (f16, row-major)
//                             @ X[8192][256] given as BT[256][8192] (f16)
// Block tile 128x64, BK=32, 8 waves (4 along M x 2 along N), wave tile 32x32
// = 4x v_wmma_f32_16x16x32_f16 per K-step. Double-buffered LDS; tiles are
// filled either by async global->LDS (ASYNCcnt) or reg staging fallback.
// ---------------------------------------------------------------------------
__global__ __launch_bounds__(256) void hg_gemm_wmma(
    const _Float16* __restrict__ A, const _Float16* __restrict__ BT,
    float* __restrict__ C, int M)
{
  constexpr int BK  = 32;
  constexpr int LDA = 40;  // padded row stride (halves): 16B-aligned, bank-spread
  __shared__ _Float16 As[2][128 * LDA];
  __shared__ _Float16 Bs[2][64 * LDA];

  const int tid  = threadIdx.x;
  const int lane = tid & 31;
  const int wave = tid >> 5;
  const int wm = wave & 3;       // M offset = 32*wm
  const int wn = wave >> 2;      // N offset = 32*wn
  const int lr = lane & 15;
  const int hh = lane >> 4;

  const int bm = blockIdx.x * 128;
  const int bn = blockIdx.y * 64;

  // global->LDS staging layout (per thread: 2x16B of A, 1x16B of B per K-step)
  const int a_row = tid >> 1;            // 0..127
  const int a_seg = (tid & 1) * 16;      // halves 0 or 16
  const int b_row = tid >> 2;            // 0..63
  const int b_seg = (tid & 3) * 8;       // halves 0,8,16,24

  const _Float16* Aptr = A  + (size_t)(bm + a_row) * KK + a_seg;
  const _Float16* Bptr = BT + (size_t)(bn + b_row) * KK + b_seg;

  f8 acc[2][2] = {};
  constexpr int KSTEPS = KK / BK;   // 256

#if defined(HG_ASYNC_LDS)
  auto issue_tile = [&](int kt, int buf) {
    const _Float16* ap = Aptr + (size_t)kt * BK;
    const _Float16* bp = Bptr + (size_t)kt * BK;
    hg_async_b128(ap,     &As[buf][a_row * LDA + a_seg]);
    hg_async_b128(ap + 8, &As[buf][a_row * LDA + a_seg + 8]);
    hg_async_b128(bp,     &Bs[buf][b_row * LDA + b_seg]);
  };
  issue_tile(0, 0);
  hg_wait_async0();
  __syncthreads();
#else
  h8 ar0, ar1, br0;
  ar0 = *(const h8*)(Aptr);
  ar1 = *(const h8*)(Aptr + 8);
  br0 = *(const h8*)(Bptr);
  *(h8*)(&As[0][a_row * LDA + a_seg])     = ar0;
  *(h8*)(&As[0][a_row * LDA + a_seg + 8]) = ar1;
  *(h8*)(&Bs[0][b_row * LDA + b_seg])     = br0;
  __syncthreads();
#endif

#pragma unroll 1
  for (int kt = 0; kt < KSTEPS; ++kt) {
    const int cur = kt & 1;

#if defined(HG_ASYNC_LDS)
    if (kt + 1 < KSTEPS) issue_tile(kt + 1, cur ^ 1);
#else
    if (kt + 1 < KSTEPS) {
      const _Float16* ap = Aptr + (size_t)(kt + 1) * BK;
      const _Float16* bp = Bptr + (size_t)(kt + 1) * BK;
      ar0 = *(const h8*)(ap);
      ar1 = *(const h8*)(ap + 8);
      br0 = *(const h8*)(bp);
      if (kt + 2 < KSTEPS) {                 // CDNA5 global_prefetch
        __builtin_prefetch(Aptr + (size_t)(kt + 2) * BK, 0, 1);
      }
    }
#endif

    // A fragment: lane holds M = lr, K chunk [16h, 16h+16)
    h16 af[2];
#pragma unroll
    for (int mt = 0; mt < 2; ++mt) {
      const int row = wm * 32 + mt * 16 + lr;
      h8 lo = *(const h8*)(&As[cur][row * LDA + 16 * hh]);
      h8 hi = *(const h8*)(&As[cur][row * LDA + 16 * hh + 8]);
      af[mt] = __builtin_shufflevector(lo, hi, 0,1,2,3,4,5,6,7,8,9,10,11,12,13,14,15);
    }
    // B fragment: lane holds N = lr, K chunk [16h, 16h+16)
    h16 bf[2];
#pragma unroll
    for (int nt = 0; nt < 2; ++nt) {
      const int row = wn * 32 + nt * 16 + lr;
      h8 lo = *(const h8*)(&Bs[cur][row * LDA + 16 * hh]);
      h8 hi = *(const h8*)(&Bs[cur][row * LDA + 16 * hh + 8]);
      bf[nt] = __builtin_shufflevector(lo, hi, 0,1,2,3,4,5,6,7,8,9,10,11,12,13,14,15);
    }

#pragma unroll
    for (int mt = 0; mt < 2; ++mt)
#pragma unroll
      for (int nt = 0; nt < 2; ++nt)
        acc[mt][nt] = __builtin_amdgcn_wmma_f32_16x16x32_f16(
            false, af[mt], false, bf[nt], (short)0, acc[mt][nt], false, false);

#if defined(HG_ASYNC_LDS)
    if (kt + 1 < KSTEPS) hg_wait_async0();
#else
    if (kt + 1 < KSTEPS) {
      const int nxt = cur ^ 1;
      *(h8*)(&As[nxt][a_row * LDA + a_seg])     = ar0;
      *(h8*)(&As[nxt][a_row * LDA + a_seg + 8]) = ar1;
      *(h8*)(&Bs[nxt][b_row * LDA + b_seg])     = br0;
    }
#endif
    __syncthreads();
  }

  // epilogue: C/D layout — VGPR r: M = r + 8*h, N = lane&15
#pragma unroll
  for (int mt = 0; mt < 2; ++mt)
#pragma unroll
    for (int nt = 0; nt < 2; ++nt) {
      const int col = bn + wn * 32 + nt * 16 + lr;
#pragma unroll
      for (int r = 0; r < 8; ++r) {
        const int row = bm + wm * 32 + mt * 16 + r + 8 * hh;
        C[(size_t)row * DD + col] = acc[mt][nt][r];
      }
    }
}

// ---------------------------------------------------------------------------
// LayerNorm over D=256 with residual: out = LN(src + emb) * gamma + beta
// ---------------------------------------------------------------------------
__global__ __launch_bounds__(256) void hg_layernorm(
    const float* __restrict__ src, const float* __restrict__ emb,
    const float* __restrict__ gamma, const float* __restrict__ beta,
    float* __restrict__ out)
{
  const int row = blockIdx.x;
  const int d   = threadIdx.x;
  const float x = src[(size_t)row * DD + d] + emb[(size_t)row * DD + d];

  float s = x, q = x * x;
#pragma unroll
  for (int off = 16; off >= 1; off >>= 1) {
    s += __shfl_xor(s, off);
    q += __shfl_xor(q, off);
  }
  __shared__ float red_s[8], red_q[8];
  const int wave = d >> 5, lane = d & 31;
  if (lane == 0) { red_s[wave] = s; red_q[wave] = q; }
  __syncthreads();
  float ts = 0.f, tq = 0.f;
#pragma unroll
  for (int i = 0; i < 8; ++i) { ts += red_s[i]; tq += red_q[i]; }
  const float mu  = ts * (1.0f / DD);
  const float var = tq * (1.0f / DD) - mu * mu;
  out[(size_t)row * DD + d] = (x - mu) * rsqrtf(var + 1e-5f) * gamma[d] + beta[d];
}

// ---------------------------------------------------------------------------
// counts[e] = sum_n Hb[n][e]; 0/1 float adds are exact -> atomic order-safe.
// ---------------------------------------------------------------------------
__global__ __launch_bounds__(256) void hg_zero(float* p, int n) {
  int i = blockIdx.x * 256 + threadIdx.x;
  if (i < n) p[i] = 0.0f;
}

__global__ __launch_bounds__(256) void hg_counts(
    const float* __restrict__ Hb, float* __restrict__ counts)
{
  const int e  = blockIdx.x * 256 + threadIdx.x;  // gridDim.x = 16
  const int n0 = blockIdx.y * 256;                // gridDim.y = 32
  float c = 0.f;
#pragma unroll 4
  for (int n = 0; n < 256; ++n) c += Hb[(size_t)(n0 + n) * EE + e];
  atomicAdd(&counts[e], c);
}

// ---------------------------------------------------------------------------
// means + max reduction
// ---------------------------------------------------------------------------
__global__ __launch_bounds__(256) void hg_partial_max(
    const float* __restrict__ sums, const float* __restrict__ counts,
    float* __restrict__ partials)
{
  const int d  = threadIdx.x;
  const int e0 = blockIdx.x * 128;   // gridDim.x = 32
  float m = -3.402823466e38f;
#pragma unroll 4
  for (int e = 0; e < 128; ++e)
    m = fmaxf(m, sums[(size_t)(e0 + e) * DD + d] / counts[e0 + e]);
  partials[blockIdx.x * DD + d] = m;
}

__global__ __launch_bounds__(256) void hg_final_max(
    const float* __restrict__ partials, float* __restrict__ out)
{
  const int d = threadIdx.x;
  float m = partials[d];
#pragma unroll
  for (int b = 1; b < 32; ++b) m = fmaxf(m, partials[b * DD + d]);
  out[d] = m;
}

// ---------------------------------------------------------------------------
// Host launcher (graph-capture safe: only kernel launches on `stream`)
// ---------------------------------------------------------------------------
extern "C" void kernel_launch(void* const* d_in, const int* in_sizes, int n_in,
                              void* d_out, int out_size, void* d_ws, size_t ws_size,
                              hipStream_t stream)
{
  (void)in_sizes; (void)n_in; (void)out_size; (void)ws_size;
  const float* emb   = (const float*)d_in[0];   // [8192][256]
  const float* A     = (const float*)d_in[1];   // [8192][8192]
  const float* Hb    = (const float*)d_in[2];   // [8192][4096]
  const float* gamma = (const float*)d_in[3];   // [256]
  const float* beta  = (const float*)d_in[4];   // [256]
  // d_in[5] = num_layers (== 3 per reference; fixed for determinism)

  char* ws = (char*)d_ws;
  auto carve = [&](size_t bytes) {
    char* p = ws;
    ws += (bytes + 255) & ~(size_t)255;
    return p;
  };
  _Float16* A_h   = (_Float16*)carve((size_t)NN * KK * 2);  // 128 MB
  _Float16* AT_h  = (_Float16*)carve((size_t)NN * KK * 2);  // 128 MB
  _Float16* HbT_h = (_Float16*)carve((size_t)EE * KK * 2);  //  64 MB
  _Float16* bT1   = (_Float16*)carve((size_t)DD * KK * 2);  //   4 MB (x^T)
  _Float16* bT2   = (_Float16*)carve((size_t)DD * KK * 2);  //   4 MB (t^T / src^T)
  float* t_buf    = (float*)carve((size_t)NN * DD * 4);     //   8 MB
  float* src_buf  = (float*)carve((size_t)NN * DD * 4);     //   8 MB
  float* last     = (float*)carve((size_t)NN * DD * 4);     //   8 MB
  float* sums     = (float*)carve((size_t)EE * DD * 4);     //   4 MB
  float* counts   = (float*)carve((size_t)EE * 4);
  float* partials = (float*)carve((size_t)32 * DD * 4);

  const dim3 blk(256);

  // one-time conversions (per call; deterministic)
  hg_transpose_cvt<<<dim3(128, 128), blk, 0, stream>>>(A,   AT_h, A_h,   NN, KK);
  hg_transpose_cvt<<<dim3(128,  64), blk, 0, stream>>>(Hb,  HbT_h, nullptr, NN, EE);
  hg_transpose_cvt<<<dim3(128,   4), blk, 0, stream>>>(emb, bT1,  nullptr, NN, DD);
  hg_zero<<<dim3(16), blk, 0, stream>>>(counts, EE);
  hg_counts<<<dim3(16, 32), blk, 0, stream>>>(Hb, counts);

  for (int layer = 0; layer < 3; ++layer) {
    hg_gemm_wmma<<<dim3(64, 4), blk, 0, stream>>>(A_h,  bT1, t_buf,   NN);
    hg_transpose_cvt<<<dim3(128, 4), blk, 0, stream>>>(t_buf, bT2, nullptr, NN, DD);
    hg_gemm_wmma<<<dim3(64, 4), blk, 0, stream>>>(AT_h, bT2, src_buf, NN);
    hg_layernorm<<<dim3(NN), blk, 0, stream>>>(src_buf, emb, gamma, beta, last);
    if (layer < 2)
      hg_transpose_cvt<<<dim3(128, 4), blk, 0, stream>>>(last, bT1, nullptr, NN, DD);
  }

  // hyperedge masked mean as WMMA GEMM: sums = Hb^T @ source
  hg_transpose_cvt<<<dim3(128, 4), blk, 0, stream>>>(src_buf, bT2, nullptr, NN, DD);
  hg_gemm_wmma<<<dim3(32, 4), blk, 0, stream>>>(HbT_h, bT2, sums, EE);
  hg_partial_max<<<dim3(32), blk, 0, stream>>>(sums, counts, partials);
  hg_final_max<<<dim3(1), blk, 0, stream>>>(partials, (float*)d_out);
}
